// FeatureFusion_81252191306399
// MI455X (gfx1250) — matrix operand order repaired and checked by reading, not compiled, not run on hardware
//
#include <hip/hip_runtime.h>
#include <hip/hip_bf16.h>

// ---------------------------------------------------------------------------
// Problem constants (from reference): B=16, SV=1024, ST=512, D=768, H=8, DH=96
// ---------------------------------------------------------------------------
#define BB      16
#define SV      1024
#define STX     512
#define DMODEL  768
#define NHEAD   8
#define DHEAD   96
#define NVROWS  (BB * SV)           // 16384 visual rows
#define NTROWS  (BB * STX)          // 8192 text rows
#define NROWS   (NVROWS + NTROWS)   // 24576 total token rows

typedef __attribute__((ext_vector_type(16))) __bf16 v16bf;
typedef __attribute__((ext_vector_type(8)))  float  v8f;
typedef int v4i_b __attribute__((vector_size(16)));   // matches builtin param type

#define GLB_AS(p) ((__attribute__((address_space(1))) v4i_b*)(p))
#define LDS_AS(p) ((__attribute__((address_space(3))) v4i_b*)(p))

__device__ __forceinline__ __bf16 f2bf(float f) {
    unsigned u = __builtin_bit_cast(unsigned, f);
    unsigned r = (u + 0x7FFFu + ((u >> 16) & 1u)) >> 16;  // round-to-nearest-even
    unsigned short h = (unsigned short)r;
    return __builtin_bit_cast(__bf16, h);
}

__device__ __forceinline__ void wait_async_then_barrier() {
    asm volatile("s_wait_asynccnt 0x0" ::: "memory");
    __syncthreads();
}

// ---------------------------------------------------------------------------
// f32 -> bf16 conversion (weights)
// ---------------------------------------------------------------------------
__global__ void cvt_bf16_kernel(const float* __restrict__ in, __bf16* __restrict__ out, int n) {
    int i = blockIdx.x * blockDim.x + threadIdx.x;
    if (i < n) out[i] = f2bf(in[i]);
}

// ---------------------------------------------------------------------------
// GEMM v2: C[NROWS,768] = concat(vis,txt) @ W + bias -> bf16 (Q/K/V).
// 256 threads (8 waves), workgroup tile 64x64, K-step 32.
// Wave w: M-tile (w&3)*16, N-tiles (w>>2)*32 and +16 (shared A fragment).
// A tile (f32 source) staged manually with convert; W tile staged with
// global_load_async_to_lds_b128 (bf16, contiguous, 16B-aligned).
// ---------------------------------------------------------------------------
__global__ __launch_bounds__(256) void gemm_qkv_kernel(
    const float* __restrict__ vis, const float* __restrict__ txt,
    const __bf16* __restrict__ W, const float* __restrict__ bias,
    __bf16* __restrict__ out)
{
    __shared__ __align__(16) __bf16 As[64][40];   // 80B row stride (16B aligned)
    __shared__ __align__(16) __bf16 Bs[32][72];   // 144B row stride (16B aligned)

    const int row0 = blockIdx.x * 64;
    const int col0 = blockIdx.y * 64;
    const int tid  = threadIdx.x;
    const int lane = tid & 31;
    const int wave = tid >> 5;
    const int m0 = (wave & 3) * 16;
    const int n0 = (wave >> 2) * 32;

    const int mA  = lane & 15;
    const int kbA = (lane < 16) ? 0 : 8;
    const int nB  = lane & 15;
    const int kbB = (lane < 16) ? 0 : 8;

    const int lrA = tid >> 2;        // 0..63 : A row
    const int lcA = (tid & 3) * 8;   // 0,8,16,24 : A col group
    const int krB = tid >> 3;        // 0..31 : W row
    const int ncB = (tid & 7) * 8;   // 0..56 : W col group

    v8f acc0 = {}, acc1 = {};

    const int r = row0 + lrA;
    const float* srcA = (r < NVROWS) ? (vis + (size_t)r * DMODEL)
                                     : (txt + (size_t)(r - NVROWS) * DMODEL);

    for (int k0 = 0; k0 < DMODEL; k0 += 32) {
        // A: f32 -> bf16 convert while staging
        #pragma unroll
        for (int i = 0; i < 8; ++i)
            As[lrA][lcA + i] = f2bf(srcA[k0 + lcA + i]);
        // B: async DMA of 16B per lane, native [k][n] layout
        __builtin_amdgcn_global_load_async_to_lds_b128(
            GLB_AS(&W[(size_t)(k0 + krB) * DMODEL + col0 + ncB]),
            LDS_AS(&Bs[krB][ncB]), 0, 0);
        if (k0 + 32 < DMODEL)
            __builtin_prefetch(&srcA[k0 + 32 + lcA], 0, 0);
        wait_async_then_barrier();

        v16bf a, b0, b1;
        #pragma unroll
        for (int v = 0; v < 8; ++v) {
            int k = (v < 4) ? (kbA + 2 * v) : (kbA + 16 + 2 * (v - 4));
            a[2 * v]     = As[m0 + mA][k];
            a[2 * v + 1] = As[m0 + mA][k + 1];
            int k2 = (v < 4) ? (kbB + 2 * v) : (kbB + 16 + 2 * (v - 4));
            b0[2 * v]     = Bs[k2][n0 + nB];
            b0[2 * v + 1] = Bs[k2 + 1][n0 + nB];
            b1[2 * v]     = Bs[k2][n0 + 16 + nB];
            b1[2 * v + 1] = Bs[k2 + 1][n0 + 16 + nB];
        }
        acc0 = __builtin_amdgcn_wmma_f32_16x16x32_bf16(false, a, false, b0,
                                                       (short)0, acc0, false, false);
        acc1 = __builtin_amdgcn_wmma_f32_16x16x32_bf16(false, a, false, b1,
                                                       (short)0, acc1, false, false);
        __syncthreads();
    }

    const int half = lane >> 4;
    const int colA = col0 + n0 + nB;
    const int colB = colA + 16;
    const float bvA = bias[colA];
    const float bvB = bias[colB];
    #pragma unroll
    for (int v = 0; v < 8; ++v) {
        size_t row = (size_t)(row0 + m0 + v + 8 * half);
        out[row * DMODEL + colA] = f2bf(acc0[v] + bvA);
        out[row * DMODEL + colB] = f2bf(acc1[v] + bvB);
    }
}

// ---------------------------------------------------------------------------
// GEMM v2: Proj[NROWS,768] = AO(bf16) @ Wo + bo -> f32 (residual+LN path).
// Both A and B tiles staged with async b128 DMA (bf16 sources).
// ---------------------------------------------------------------------------
__global__ __launch_bounds__(256) void gemm_out_kernel(
    const __bf16* __restrict__ A,
    const __bf16* __restrict__ W, const float* __restrict__ bias,
    float* __restrict__ out)
{
    __shared__ __align__(16) __bf16 As[64][40];
    __shared__ __align__(16) __bf16 Bs[32][72];

    const int row0 = blockIdx.x * 64;
    const int col0 = blockIdx.y * 64;
    const int tid  = threadIdx.x;
    const int lane = tid & 31;
    const int wave = tid >> 5;
    const int m0 = (wave & 3) * 16;
    const int n0 = (wave >> 2) * 32;

    const int mA  = lane & 15;
    const int kbA = (lane < 16) ? 0 : 8;
    const int nB  = lane & 15;
    const int kbB = (lane < 16) ? 0 : 8;

    const int lrA = tid >> 2;
    const int lcA = (tid & 3) * 8;
    const int krB = tid >> 3;
    const int ncB = (tid & 7) * 8;

    v8f acc0 = {}, acc1 = {};
    const __bf16* srcA = A + (size_t)(row0 + lrA) * DMODEL;

    for (int k0 = 0; k0 < DMODEL; k0 += 32) {
        __builtin_amdgcn_global_load_async_to_lds_b128(
            GLB_AS(&srcA[k0 + lcA]), LDS_AS(&As[lrA][lcA]), 0, 0);
        __builtin_amdgcn_global_load_async_to_lds_b128(
            GLB_AS(&W[(size_t)(k0 + krB) * DMODEL + col0 + ncB]),
            LDS_AS(&Bs[krB][ncB]), 0, 0);
        wait_async_then_barrier();

        v16bf a, b0, b1;
        #pragma unroll
        for (int v = 0; v < 8; ++v) {
            int k = (v < 4) ? (kbA + 2 * v) : (kbA + 16 + 2 * (v - 4));
            a[2 * v]     = As[m0 + mA][k];
            a[2 * v + 1] = As[m0 + mA][k + 1];
            int k2 = (v < 4) ? (kbB + 2 * v) : (kbB + 16 + 2 * (v - 4));
            b0[2 * v]     = Bs[k2][n0 + nB];
            b0[2 * v + 1] = Bs[k2 + 1][n0 + nB];
            b1[2 * v]     = Bs[k2][n0 + 16 + nB];
            b1[2 * v + 1] = Bs[k2 + 1][n0 + 16 + nB];
        }
        acc0 = __builtin_amdgcn_wmma_f32_16x16x32_bf16(false, a, false, b0,
                                                       (short)0, acc0, false, false);
        acc1 = __builtin_amdgcn_wmma_f32_16x16x32_bf16(false, a, false, b1,
                                                       (short)0, acc1, false, false);
        __syncthreads();
    }

    const int half = lane >> 4;
    const int colA = col0 + n0 + nB;
    const int colB = colA + 16;
    const float bvA = bias[colA];
    const float bvB = bias[colB];
    #pragma unroll
    for (int v = 0; v < 8; ++v) {
        size_t row = (size_t)(row0 + m0 + v + 8 * half);
        out[row * DMODEL + colA] = acc0[v] + bvA;
        out[row * DMODEL + colB] = acc1[v] + bvB;
    }
}

// ---------------------------------------------------------------------------
// Flash cross-attention. One wave per (b, h, 16-query tile).
// Scores: Q(16x96) @ K^T via 3 WMMA per 16-key column tile (2 tiles / chunk).
// Online softmax with __shfl_xor row reductions; P restaged via LDS into
// A-fragment layout; out += P(16x32) @ V(32x96) via 6 WMMA per chunk.
// ---------------------------------------------------------------------------
__global__ __launch_bounds__(32) void attn_kernel(
    const __bf16* __restrict__ Qb, const __bf16* __restrict__ Kb,
    const __bf16* __restrict__ Vb, const int* __restrict__ kv_mask,
    __bf16* __restrict__ Ob,
    int qbase, int kbase, int Sq, int Sk)
{
    const int b  = blockIdx.z;
    const int h  = blockIdx.y;
    const int qt = blockIdx.x;
    const int lane = threadIdx.x & 31;
    const int half = lane >> 4;
    const int nl   = lane & 15;

    const long qrow0 = (long)qbase + (long)b * Sq + (long)qt * 16;
    const long krow0 = (long)kbase + (long)b * Sk;
    const int  col0  = h * DHEAD;

    const int kbA = half ? 8 : 0;    // A-fragment K base
    const int kbB = half ? 8 : 0;    // B-fragment K base
    const int m   = nl;              // A-fragment row

    // Preload Q A-fragments (3 segments of K=32 covering DH=96)
    v16bf aq[3];
    #pragma unroll
    for (int seg = 0; seg < 3; ++seg) {
        #pragma unroll
        for (int v = 0; v < 8; ++v) {
            int k = (v < 4) ? (kbA + 2 * v) : (kbA + 16 + 2 * (v - 4));
            const __bf16* p = Qb + (qrow0 + m) * DMODEL + col0 + seg * 32 + k;
            aq[seg][2 * v]     = p[0];
            aq[seg][2 * v + 1] = p[1];
        }
    }

    v8f o[6];
    #pragma unroll
    for (int t = 0; t < 6; ++t) o[t] = (v8f){};
    float rmax[8], rsum[8];
    #pragma unroll
    for (int v = 0; v < 8; ++v) { rmax[v] = -1e30f; rsum[v] = 0.0f; }

    __shared__ __bf16 Pl[16 * 32];
    const float scale = 0.10206207262f;   // 1/sqrt(96)
    const int* maskp = kv_mask + (long)b * Sk;

    for (int c0 = 0; c0 < Sk; c0 += 32) {
        // ---- scores S = Q @ K^T for 32 keys (two 16-wide column tiles) ----
        v8f s0 = {}, s1 = {};
        #pragma unroll
        for (int seg = 0; seg < 3; ++seg) {
            v16bf bk0, bk1;
            #pragma unroll
            for (int v = 0; v < 8; ++v) {
                int k = (v < 4) ? (kbB + 2 * v) : (kbB + 16 + 2 * (v - 4));   // dh offset within segment
                const __bf16* p0 = Kb + (krow0 + c0 + nl) * DMODEL + col0 + seg * 32 + k;
                bk0[2 * v]     = p0[0];
                bk0[2 * v + 1] = p0[1];
                const __bf16* p1 = Kb + (krow0 + c0 + 16 + nl) * DMODEL + col0 + seg * 32 + k;
                bk1[2 * v]     = p1[0];
                bk1[2 * v + 1] = p1[1];
            }
            s0 = __builtin_amdgcn_wmma_f32_16x16x32_bf16(false, aq[seg], false, bk0,
                                                         (short)0, s0, false, false);
            s1 = __builtin_amdgcn_wmma_f32_16x16x32_bf16(false, aq[seg], false, bk1,
                                                         (short)0, s1, false, false);
        }

        // ---- scale + mask + chunk row-max ----
        const int msk0 = maskp[c0 + nl];
        const int msk1 = maskp[c0 + 16 + nl];
        float cm[8];
        #pragma unroll
        for (int v = 0; v < 8; ++v) {
            float a0 = s0[v] * scale; if (msk0 == 0) a0 = -1e30f;
            float a1 = s1[v] * scale; if (msk1 == 0) a1 = -1e30f;
            s0[v] = a0; s1[v] = a1;
            cm[v] = fmaxf(a0, a1);
        }
        #pragma unroll
        for (int off = 8; off > 0; off >>= 1) {
            #pragma unroll
            for (int v = 0; v < 8; ++v)
                cm[v] = fmaxf(cm[v], __shfl_xor(cm[v], off, 32));
        }

        // ---- online softmax update ----
        float nm[8], corr[8], rs[8];
        #pragma unroll
        for (int v = 0; v < 8; ++v) {
            nm[v]   = fmaxf(rmax[v], cm[v]);
            corr[v] = __expf(rmax[v] - nm[v]);
            rmax[v] = nm[v];
        }
        float ps0[8], ps1[8];
        #pragma unroll
        for (int v = 0; v < 8; ++v) {
            ps0[v] = __expf(s0[v] - nm[v]);
            ps1[v] = __expf(s1[v] - nm[v]);
            rs[v]  = ps0[v] + ps1[v];
        }
        #pragma unroll
        for (int off = 8; off > 0; off >>= 1) {
            #pragma unroll
            for (int v = 0; v < 8; ++v)
                rs[v] += __shfl_xor(rs[v], off, 32);
        }
        #pragma unroll
        for (int v = 0; v < 8; ++v)
            rsum[v] = rsum[v] * corr[v] + rs[v];
        #pragma unroll
        for (int t = 0; t < 6; ++t)
            #pragma unroll
            for (int v = 0; v < 8; ++v)
                o[t][v] *= corr[v];

        // ---- restage P (C/D layout) -> A-fragment layout via LDS ----
        #pragma unroll
        for (int v = 0; v < 8; ++v) {
            int mrow = v + 8 * half;
            Pl[mrow * 32 + nl]      = f2bf(ps0[v]);
            Pl[mrow * 32 + 16 + nl] = f2bf(ps1[v]);
        }
        __syncthreads();   // single-wave WG: S_NOP, LDS ordering via DScnt
        v16bf ap;
        #pragma unroll
        for (int v = 0; v < 8; ++v) {
            int k = (v < 4) ? (kbA + 2 * v) : (kbA + 16 + 2 * (v - 4));
            ap[2 * v]     = Pl[m * 32 + k];
            ap[2 * v + 1] = Pl[m * 32 + k + 1];
        }

        // ---- O += P @ V (6 column tiles of 16 over DH=96) ----
        #pragma unroll
        for (int t = 0; t < 6; ++t) {
            v16bf bv;
            #pragma unroll
            for (int v = 0; v < 8; ++v) {
                int k = (v < 4) ? (kbB + 2 * v) : (kbB + 16 + 2 * (v - 4));   // key index within chunk
                bv[2 * v]     = Vb[(krow0 + c0 + k)     * DMODEL + col0 + t * 16 + nl];
                bv[2 * v + 1] = Vb[(krow0 + c0 + k + 1) * DMODEL + col0 + t * 16 + nl];
            }
            o[t] = __builtin_amdgcn_wmma_f32_16x16x32_bf16(false, ap, false, bv,
                                                           (short)0, o[t], false, false);
        }
        __syncthreads();
    }

    // ---- normalize and store (bf16 for the Wo GEMM) ----
    #pragma unroll
    for (int v = 0; v < 8; ++v) {
        float inv = 1.0f / rsum[v];
        long row = qrow0 + v + 8 * half;
        #pragma unroll
        for (int t = 0; t < 6; ++t)
            Ob[row * DMODEL + col0 + t * 16 + nl] = f2bf(o[t][v] * inv);
    }
}

// ---------------------------------------------------------------------------
// Residual + LayerNorm + masked pooled accumulation. One block per token row.
// ---------------------------------------------------------------------------
__global__ __launch_bounds__(256) void lnpool_kernel(
    const float* __restrict__ vis, const float* __restrict__ txt,
    const float* __restrict__ proj,
    const int* __restrict__ vmask, const int* __restrict__ tmask,
    const float* __restrict__ g, const float* __restrict__ bta,
    float* __restrict__ pooled, float* __restrict__ counts)
{
    const long r = blockIdx.x;
    const int tid = threadIdx.x;
    const float* x = (r < NVROWS) ? (vis + r * DMODEL) : (txt + (r - NVROWS) * DMODEL);
    const float* y = proj + r * DMODEL;

    __shared__ float rowbuf[DMODEL];
    __shared__ float red[256];

    float s = 0.0f;
    for (int j = tid; j < DMODEL; j += 256) {
        float v = x[j] + y[j];
        rowbuf[j] = v;
        s += v;
    }
    red[tid] = s; __syncthreads();
    for (int o = 128; o > 0; o >>= 1) { if (tid < o) red[tid] += red[tid + o]; __syncthreads(); }
    const float mean = red[0] / (float)DMODEL;
    __syncthreads();

    float s2 = 0.0f;
    for (int j = tid; j < DMODEL; j += 256) {
        float d = rowbuf[j] - mean;
        s2 += d * d;
    }
    red[tid] = s2; __syncthreads();
    for (int o = 128; o > 0; o >>= 1) { if (tid < o) red[tid] += red[tid + o]; __syncthreads(); }
    const float inv = rsqrtf(red[0] / (float)DMODEL + 1e-5f);

    int b, seg, mask;
    if (r < NVROWS) { b = (int)(r / SV);            seg = 0; mask = vmask[r]; }
    else            { long rr = r - NVROWS; b = (int)(rr / STX); seg = 1; mask = tmask[rr]; }

    if (mask != 0) {
        float* dst = pooled + (size_t)b * (2 * DMODEL) + (size_t)seg * DMODEL;
        for (int j = tid; j < DMODEL; j += 256) {
            float v = (rowbuf[j] - mean) * inv * g[j] + bta[j];
            atomicAdd(&dst[j], v);
        }
        if (tid == 0) atomicAdd(&counts[b * 2 + seg], 1.0f);
    }
}

// ---------------------------------------------------------------------------
// Final fusion: [B,1536] @ Wf + bf, then LayerNorm -> d_out [B,768] (f32).
// One block per batch row (tiny GEMM, keep f32 VALU for precision).
// ---------------------------------------------------------------------------
__global__ __launch_bounds__(256) void fuse_kernel(
    const float* __restrict__ pooled, const float* __restrict__ counts,
    const float* __restrict__ Wf, const float* __restrict__ bf_,
    const float* __restrict__ g2, const float* __restrict__ b2,
    float* __restrict__ out)
{
    const int b = blockIdx.x;
    const int tid = threadIdx.x;
    __shared__ float pv[2 * DMODEL];
    __shared__ float fbuf[DMODEL];
    __shared__ float red[256];

    const float c0 = counts[b * 2 + 0];
    const float c1 = counts[b * 2 + 1];
    const float i0 = (c0 > 0.0f) ? (1.0f / c0) : 0.0f;
    const float i1 = (c1 > 0.0f) ? (1.0f / c1) : 0.0f;

    for (int i = tid; i < 2 * DMODEL; i += 256)
        pv[i] = pooled[(size_t)b * (2 * DMODEL) + i] * ((i < DMODEL) ? i0 : i1);
    __syncthreads();

    for (int j = tid; j < DMODEL; j += 256) {
        float acc = bf_[j];
        for (int i = 0; i < 2 * DMODEL; ++i)
            acc += pv[i] * Wf[(size_t)i * DMODEL + j];
        fbuf[j] = acc;
    }
    __syncthreads();

    float s = 0.0f;
    for (int j = tid; j < DMODEL; j += 256) s += fbuf[j];
    red[tid] = s; __syncthreads();
    for (int o = 128; o > 0; o >>= 1) { if (tid < o) red[tid] += red[tid + o]; __syncthreads(); }
    const float mean = red[0] / (float)DMODEL;
    __syncthreads();

    float s2 = 0.0f;
    for (int j = tid; j < DMODEL; j += 256) { float d = fbuf[j] - mean; s2 += d * d; }
    red[tid] = s2; __syncthreads();
    for (int o = 128; o > 0; o >>= 1) { if (tid < o) red[tid] += red[tid + o]; __syncthreads(); }
    const float inv = rsqrtf(red[0] / (float)DMODEL + 1e-5f);

    for (int j = tid; j < DMODEL; j += 256)
        out[(size_t)b * DMODEL + j] = (fbuf[j] - mean) * inv * g2[j] + b2[j];
}

// ---------------------------------------------------------------------------
// Host launch
// ---------------------------------------------------------------------------
extern "C" void kernel_launch(void* const* d_in, const int* in_sizes, int n_in,
                              void* d_out, int out_size, void* d_ws, size_t ws_size,
                              hipStream_t stream)
{
    const float* vis   = (const float*)d_in[0];
    const float* txt   = (const float*)d_in[1];
    const int*   vmask = (const int*)d_in[2];
    const int*   tmask = (const int*)d_in[3];
    const float* Wq  = (const float*)d_in[4];
    const float* bq  = (const float*)d_in[5];
    const float* Wk  = (const float*)d_in[6];
    const float* bk  = (const float*)d_in[7];
    const float* Wv  = (const float*)d_in[8];
    const float* bv  = (const float*)d_in[9];
    const float* Wo  = (const float*)d_in[10];
    const float* bo  = (const float*)d_in[11];
    const float* g1  = (const float*)d_in[12];
    const float* b1  = (const float*)d_in[13];
    const float* Wf  = (const float*)d_in[14];
    const float* bfv = (const float*)d_in[15];
    const float* g2  = (const float*)d_in[16];
    const float* b2  = (const float*)d_in[17];

    // workspace carve-up (256-byte aligned)
    size_t off = 0;
    char* base = (char*)d_ws;
    auto carve = [&](size_t bytes) -> void* {
        void* p = base + off;
        off += (bytes + 255) & ~(size_t)255;
        return p;
    };
    const size_t tokBytesBf = (size_t)NROWS * DMODEL * sizeof(__bf16);
    __bf16* Qb  = (__bf16*)carve(tokBytesBf);
    __bf16* Kb  = (__bf16*)carve(tokBytesBf);
    __bf16* Vb  = (__bf16*)carve(tokBytesBf);
    __bf16* AOb = (__bf16*)carve(tokBytesBf);
    float*  Proj = (float*)carve((size_t)NROWS * DMODEL * sizeof(float));
    const size_t wBytes = (size_t)DMODEL * DMODEL * sizeof(__bf16);
    __bf16* Wqb = (__bf16*)carve(wBytes);
    __bf16* Wkb = (__bf16*)carve(wBytes);
    __bf16* Wvb = (__bf16*)carve(wBytes);
    __bf16* Wob = (__bf16*)carve(wBytes);
    float* pooled = (float*)carve((size_t)BB * 2 * DMODEL * sizeof(float));
    float* counts = (float*)carve((size_t)BB * 2 * sizeof(float));

    // 1) weights -> bf16
    const int wN = DMODEL * DMODEL;
    cvt_bf16_kernel<<<(wN + 255) / 256, 256, 0, stream>>>(Wq, Wqb, wN);
    cvt_bf16_kernel<<<(wN + 255) / 256, 256, 0, stream>>>(Wk, Wkb, wN);
    cvt_bf16_kernel<<<(wN + 255) / 256, 256, 0, stream>>>(Wv, Wvb, wN);
    cvt_bf16_kernel<<<(wN + 255) / 256, 256, 0, stream>>>(Wo, Wob, wN);

    // 2) QKV projections (shared weights over all 24576 token rows)
    dim3 ggrid(NROWS / 64, DMODEL / 64);
    gemm_qkv_kernel<<<ggrid, 256, 0, stream>>>(vis, txt, Wqb, bq, Qb);
    gemm_qkv_kernel<<<ggrid, 256, 0, stream>>>(vis, txt, Wkb, bk, Kb);
    gemm_qkv_kernel<<<ggrid, 256, 0, stream>>>(vis, txt, Wvb, bv, Vb);

    // 3) cross-attention, both directions
    attn_kernel<<<dim3(SV / 16, NHEAD, BB), 32, 0, stream>>>(
        Qb, Kb, Vb, tmask, AOb, /*qbase=*/0, /*kbase=*/NVROWS, SV, STX);
    attn_kernel<<<dim3(STX / 16, NHEAD, BB), 32, 0, stream>>>(
        Qb, Kb, Vb, vmask, AOb, /*qbase=*/NVROWS, /*kbase=*/0, STX, SV);

    // 4) output projection (f32 out for residual + LN)
    gemm_out_kernel<<<ggrid, 256, 0, stream>>>(AOb, Wob, bo, Proj);

    // 5) residual + LN + masked pooling (zero accumulators first)
    (void)hipMemsetAsync(pooled, 0, (size_t)BB * 2 * DMODEL * sizeof(float), stream);
    (void)hipMemsetAsync(counts, 0, (size_t)BB * 2 * sizeof(float), stream);
    lnpool_kernel<<<NROWS, 256, 0, stream>>>(vis, txt, Proj, vmask, tmask,
                                             g1, b1, pooled, counts);

    // 6) fusion GEMM + final LN -> d_out [16,768] f32
    fuse_kernel<<<BB, 256, 0, stream>>>(pooled, counts, Wf, bfv, g2, b2,
                                        (float*)d_out);
}
